// ChemPropMsgToNode_5325759447401
// MI455X (gfx1250) — matrix operand order repaired and checked by reading, not compiled, not run on hardware
//
#include <hip/hip_runtime.h>
#include <hip/hip_bf16.h>

// ---------------------------------------------------------------------------
// ChemProp msg-to-node + 2-layer MLP, fused for MI455X (gfx1250, wave32).
//   msg = segment_sum(h, nbrs[:,0]);  out = relu([r,msg] @ W1) @ W2
// CSR-gather (819MB streamed once, ~35us floor at 23.3TB/s) instead of 205M
// fp32 atomics; GEMMs as v_wmma_f32_16x16x32_f16 with fp32 accumulation.
// This revision: 32 nodes per workgroup (2 M-tiles per wave) so every B
// fragment feeds 2 WMMAs -> halves L2 weight traffic (~1.2GB total) and
// halves load-latency exposure per WMMA; A-tile and hidden-tile share one
// 32KB LDS buffer (all A reads complete before hidden writes) to keep
// ~10 WGs/WGP resident for latency hiding.
// ---------------------------------------------------------------------------

typedef _Float16 v16h __attribute__((ext_vector_type(16)));
typedef _Float16 v8h  __attribute__((ext_vector_type(8)));
typedef float    v8f  __attribute__((ext_vector_type(8)));

#define FDIM   256   // node / edge feature dim
#define TWOF   512   // concat dim = GEMM1 K
#define HDIM   512   // hidden dim  = GEMM1 N = GEMM2 K
#define FOUT   256   // output dim  = GEMM2 N
#define MTILE  32    // nodes per workgroup

// ---------------------------- small prep kernels ---------------------------

__global__ void zero_i32_kernel(int* __restrict__ p, int n) {
    int i = blockIdx.x * blockDim.x + threadIdx.x;
    if (i < n) p[i] = 0;
}

__global__ void f32_to_f16_kernel(const float* __restrict__ in,
                                  _Float16* __restrict__ out, int n) {
    int i = blockIdx.x * blockDim.x + threadIdx.x;
    if (i < n) out[i] = (_Float16)in[i];
}

__global__ void edge_hist_kernel(const int* __restrict__ nbrs,
                                 int* __restrict__ deg, int n_edges) {
    int e = blockIdx.x * blockDim.x + threadIdx.x;
    if (e < n_edges) {
        int dst = nbrs[2 * e];
        atomicAdd(&deg[dst], 1);
    }
}

// single-block exclusive scan over deg -> rowptr (n+1) and cursor copy
__global__ __launch_bounds__(256) void scan_kernel(const int* __restrict__ deg,
                                                   int* __restrict__ rowptr,
                                                   int* __restrict__ cursor,
                                                   int n) {
    __shared__ int part[256];
    __shared__ int total;
    const int t  = threadIdx.x;
    const int ch = (n + 255) / 256;
    const int beg = t * ch;
    const int end = (beg + ch < n) ? (beg + ch) : n;
    int s = 0;
    for (int i = beg; i < end; ++i) s += deg[i];
    part[t] = s;
    __syncthreads();
    if (t == 0) {
        int run = 0;
        for (int i = 0; i < 256; ++i) { int v = part[i]; part[i] = run; run += v; }
        total = run;
    }
    __syncthreads();
    int run = part[t];
    for (int i = beg; i < end; ++i) {
        rowptr[i] = run;
        cursor[i] = run;
        run += deg[i];
    }
    if (t == 0) rowptr[n] = total;
}

__global__ void edge_fill_kernel(const int* __restrict__ nbrs,
                                 int* __restrict__ cursor,
                                 int* __restrict__ eidx, int n_edges) {
    int e = blockIdx.x * blockDim.x + threadIdx.x;
    if (e < n_edges) {
        int dst = nbrs[2 * e];
        int pos = atomicAdd(&cursor[dst], 1);
        eidx[pos] = e;
    }
}

// ------------------------- fused gather + MLP kernel -----------------------
// One workgroup = 128 threads (4 waves) handles 32 nodes (2 M-tiles).
// LDS Tile is reused: concat A-tile during GEMM1, hidden tile during GEMM2.
__global__ __launch_bounds__(128, 1) void chemprop_mlp_kernel(
    const float* __restrict__ r, const float* __restrict__ h,
    const int* __restrict__ rowptr, const int* __restrict__ eidx,
    const _Float16* __restrict__ W1h, const _Float16* __restrict__ W2h,
    float* __restrict__ out, int n_nodes) {

    __shared__ _Float16 Tile[MTILE * TWOF];   // 32 KB, aliased A/H tile

    const int tid  = threadIdx.x;
    const int lane = tid & 31;
    const int wave = tid >> 5;
    const int nb   = blockIdx.x * MTILE;

    // ---- phase 1: gather (segment sum) + concat; all 128 threads per node,
    //      2 feature columns per thread, edge loop unrolled x4 for MLP ----
    const int c = tid * 2;
    for (int m = 0; m < MTILE; ++m) {
        int node = nb + m;
        int nclamp = node < n_nodes ? node : n_nodes - 1;
        float a0 = 0.f, a1 = 0.f;
        const int beg = rowptr[nclamp];
        const int end = rowptr[nclamp + 1];
        int j = beg;
        for (; j + 4 <= end; j += 4) {          // 4 edge rows in flight
            const int4 e4 = *(const int4*)(eidx + j);
            const float2 h0 = *(const float2*)(h + (size_t)e4.x * FDIM + c);
            const float2 h1 = *(const float2*)(h + (size_t)e4.y * FDIM + c);
            const float2 h2 = *(const float2*)(h + (size_t)e4.z * FDIM + c);
            const float2 h3 = *(const float2*)(h + (size_t)e4.w * FDIM + c);
            a0 += h0.x + h1.x + h2.x + h3.x;
            a1 += h0.y + h1.y + h2.y + h3.y;
        }
        for (; j < end; ++j) {
            const int e = eidx[j];
            const float2 hv = *(const float2*)(h + (size_t)e * FDIM + c);
            a0 += hv.x; a1 += hv.y;
        }
        const float2 rv = *(const float2*)(r + (size_t)nclamp * FDIM + c);
        Tile[m * TWOF + c]            = (_Float16)rv.x;
        Tile[m * TWOF + c + 1]        = (_Float16)rv.y;
        Tile[m * TWOF + FDIM + c]     = (_Float16)a0;
        Tile[m * TWOF + FDIM + c + 1] = (_Float16)a1;
    }
    __syncthreads();

    // WMMA fragment coordinates (16-bit A 16x32 layout, ISA 7.12.2):
    //   lane L: row-in-tile = L&15, K-half kh = L>>4
    //   a = { A[m][ks+8*kh .. +7], A[m][ks+16+8*kh .. +7] }  (two b128 loads)
    //   B lane n = L&15; elem s: k = ks+8*kh+(s&7)+16*(s>>3) as in A; 16 f16 down one weight column
    //   C/D: row = vg + 8*kh, col = L&15
    const int mrow = lane & 15;
    const int kh   = lane >> 4;

    // ---- phase 2: GEMM1 (each wave: 2 M-tiles x 8 N-tiles) ----
    v8f acc[2][8] = {};
    for (int ks = 0; ks < TWOF; ks += 32) {
        const int kg = ks + mrow + 16 * kh;
        // A fragments for both M-tiles
        const v8h a0l = *(const v8h*)&Tile[(0 * 16 + mrow) * TWOF + ks + 8 * kh];
        const v8h a0h = *(const v8h*)&Tile[(0 * 16 + mrow) * TWOF + ks + 16 + 8 * kh];
        const v8h a1l = *(const v8h*)&Tile[(1 * 16 + mrow) * TWOF + ks + 8 * kh];
        const v8h a1h = *(const v8h*)&Tile[(1 * 16 + mrow) * TWOF + ks + 16 + 8 * kh];
        const v16h a0 = __builtin_shufflevector(a0l, a0h,
            0, 1, 2, 3, 4, 5, 6, 7, 8, 9, 10, 11, 12, 13, 14, 15);
        const v16h a1 = __builtin_shufflevector(a1l, a1h,
            0, 1, 2, 3, 4, 5, 6, 7, 8, 9, 10, 11, 12, 13, 14, 15);
#pragma unroll
        for (int t = 0; t < 8; ++t) {
            v16h b;
#pragma unroll
            for (int s = 0; s < 16; ++s)
                b[s] = W1h[(size_t)(ks + 8 * kh + (s & 7) + 16 * (s >> 3)) * HDIM + (wave * 8 + t) * 16 + mrow];
            acc[0][t] = __builtin_amdgcn_wmma_f32_16x16x32_f16(
                false, a0, false, b, (short)0, acc[0][t], false, false);
            acc[1][t] = __builtin_amdgcn_wmma_f32_16x16x32_f16(
                false, a1, false, b, (short)0, acc[1][t], false, false);
        }
    }
    __syncthreads();   // all A reads done -> safe to overwrite Tile w/ hidden

    // ---- ReLU + hidden tile (f16) into the same LDS buffer ----
#pragma unroll
    for (int t = 0; t < 8; ++t) {
        const int n = (wave * 8 + t) * 16 + (lane & 15);
#pragma unroll
        for (int mt = 0; mt < 2; ++mt) {
#pragma unroll
            for (int vg = 0; vg < 8; ++vg) {
                const int m = mt * 16 + vg + 8 * kh;
                float x = acc[mt][t][vg];
                Tile[m * HDIM + n] = (_Float16)(x > 0.f ? x : 0.f);
            }
        }
    }
    __syncthreads();

    // ---- phase 3: GEMM2 (each wave: 2 M-tiles x 4 N-tiles) ----
    v8f acc2[2][4] = {};
    for (int ks = 0; ks < HDIM; ks += 32) {
        const int kg = ks + mrow + 16 * kh;
        const v8h a0l = *(const v8h*)&Tile[(0 * 16 + mrow) * HDIM + ks + 8 * kh];
        const v8h a0h = *(const v8h*)&Tile[(0 * 16 + mrow) * HDIM + ks + 16 + 8 * kh];
        const v8h a1l = *(const v8h*)&Tile[(1 * 16 + mrow) * HDIM + ks + 8 * kh];
        const v8h a1h = *(const v8h*)&Tile[(1 * 16 + mrow) * HDIM + ks + 16 + 8 * kh];
        const v16h a0 = __builtin_shufflevector(a0l, a0h,
            0, 1, 2, 3, 4, 5, 6, 7, 8, 9, 10, 11, 12, 13, 14, 15);
        const v16h a1 = __builtin_shufflevector(a1l, a1h,
            0, 1, 2, 3, 4, 5, 6, 7, 8, 9, 10, 11, 12, 13, 14, 15);
#pragma unroll
        for (int t = 0; t < 4; ++t) {
            v16h b;
#pragma unroll
            for (int s = 0; s < 16; ++s)
                b[s] = W2h[(size_t)(ks + 8 * kh + (s & 7) + 16 * (s >> 3)) * FOUT + (wave * 4 + t) * 16 + mrow];
            acc2[0][t] = __builtin_amdgcn_wmma_f32_16x16x32_f16(
                false, a0, false, b, (short)0, acc2[0][t], false, false);
            acc2[1][t] = __builtin_amdgcn_wmma_f32_16x16x32_f16(
                false, a1, false, b, (short)0, acc2[1][t], false, false);
        }
    }

    // ---- fp32 output stores ----
#pragma unroll
    for (int t = 0; t < 4; ++t) {
        const int n = (wave * 4 + t) * 16 + (lane & 15);
#pragma unroll
        for (int mt = 0; mt < 2; ++mt) {
#pragma unroll
            for (int vg = 0; vg < 8; ++vg) {
                const int m = mt * 16 + vg + 8 * kh;
                if (nb + m < n_nodes)
                    out[(size_t)(nb + m) * FOUT + n] = acc2[mt][t][vg];
            }
        }
    }
}

// --------------------------------- launch ----------------------------------

extern "C" void kernel_launch(void* const* d_in, const int* in_sizes, int n_in,
                              void* d_out, int out_size, void* d_ws, size_t ws_size,
                              hipStream_t stream) {
    const float* r    = (const float*)d_in[0];
    const float* h    = (const float*)d_in[1];
    const int*   nbrs = (const int*)d_in[2];
    const float* W1   = (const float*)d_in[3];
    const float* W2   = (const float*)d_in[4];
    float*       out  = (float*)d_out;

    const int n_nodes = in_sizes[0] / FDIM;   // 50000
    const int n_edges = in_sizes[1] / FDIM;   // 800000

    // workspace layout (all offsets 64B-aligned)
    char* ws = (char*)d_ws;
    _Float16* W1h   = (_Float16*)(ws + 0);           // 512*512*2 = 524288
    _Float16* W2h   = (_Float16*)(ws + 524288);      // 512*256*2 = 262144
    int* deg    = (int*)(ws + 786432);               // n_nodes*4
    int* rowptr = (int*)(ws + 986432);               // (n_nodes+1)*4
    int* cursor = (int*)(ws + 1186560);              // n_nodes*4
    int* eidx   = (int*)(ws + 1386560);              // n_edges*4

    const int TB = 256;
    zero_i32_kernel<<<(n_nodes + TB - 1) / TB, TB, 0, stream>>>(deg, n_nodes);
    f32_to_f16_kernel<<<(TWOF * HDIM + TB - 1) / TB, TB, 0, stream>>>(W1, W1h, TWOF * HDIM);
    f32_to_f16_kernel<<<(HDIM * FOUT + TB - 1) / TB, TB, 0, stream>>>(W2, W2h, HDIM * FOUT);
    edge_hist_kernel<<<(n_edges + TB - 1) / TB, TB, 0, stream>>>(nbrs, deg, n_edges);
    scan_kernel<<<1, 256, 0, stream>>>(deg, rowptr, cursor, n_nodes);
    edge_fill_kernel<<<(n_edges + TB - 1) / TB, TB, 0, stream>>>(nbrs, cursor, eidx, n_edges);

    const int n_tiles = (n_nodes + MTILE - 1) / MTILE;   // 1563
    chemprop_mlp_kernel<<<n_tiles, 128, 0, stream>>>(r, h, rowptr, eidx,
                                                     W1h, W2h, out, n_nodes);
}
